// Seq2Seq_618475290657
// MI455X (gfx1250) — compile-verified, matrix operand order checked by reading, not run on hardware
//
#include <hip/hip_runtime.h>
#include <hip/hip_bf16.h>
#include <math.h>

// ---------------------------------------------------------------------------
// Seq2Seq GRU + attention for MI455X (gfx1250).
// All GEMMs via v_wmma_f32_16x16x32_bf16 (bf16 in, fp32 accumulate).
// Each wave computes a 32x64 output tile (2 m-tiles x 4 n-tiles) so every
// B fragment is reused by two A fragments. K is a template parameter so the
// four B-row addresses collapse onto ONE base VGPR pair + immediate offsets
// (avoids the register spills seen with 6 live 64-bit pointers).
// ---------------------------------------------------------------------------

typedef __attribute__((ext_vector_type(16))) __bf16 v16bf;
typedef __attribute__((ext_vector_type(8)))  float  v8f;

union FragBF {
  v16bf v;
  uint4 q[2];
};

__device__ __forceinline__ unsigned short f2bf(float f) {
  unsigned int u = __float_as_uint(f);
  u += 0x7FFFu + ((u >> 16) & 1u);           // round-to-nearest-even
  return (unsigned short)(u >> 16);
}

// ---------------------------------------------------------------------------
// fp32 -> bf16 elementwise convert
__global__ void f32_to_bf16_k(const float* __restrict__ in,
                              unsigned short* __restrict__ out, long n) {
  long i = (long)blockIdx.x * blockDim.x + threadIdx.x;
  if (i < n) out[i] = f2bf(in[i]);
}

// bil_W [H,K] -> transposed bf16 [K,H] (so GEMM computes q = h @ W)
__global__ void transpose_bf16_k(const float* __restrict__ in,
                                 unsigned short* __restrict__ out) {
  int i = blockIdx.x * blockDim.x + threadIdx.x; // 512*512
  int n = i >> 9, k = i & 511;
  out[(long)n * 512 + k] = f2bf(in[(long)k * 512 + n]);
}

// zero hidden state, zero xh staging (bf16), seed tokens with SOS
__global__ void init_state_k(float* __restrict__ h, unsigned short* __restrict__ xh,
                             const int* __restrict__ sos, int* __restrict__ tok) {
  int i = blockIdx.x * blockDim.x + threadIdx.x;
  if (i < 128 * 512)  h[i] = 0.0f;
  if (i < 128 * 1024) xh[i] = 0;
  if (i < 128)        tok[i] = sos[0];
}

// ---------------------------------------------------------------------------
// WMMA GEMM:  C[M,N] = A[M,K] * B[N,K]^T (+ bias[N])
//   A: bf16 row-major, row stride lda (elements). Optional rowidx gather
//      (embedding-table-as-A fusion).
//   B: bf16 row-major [N,K] (PyTorch weight convention).
//   One wave computes a 32x64 strip: 2 A frags x 4 B frags -> 8 WMMAs/k-step.
//   blockDim = 256 (8 waves). Requires M%32==0, N%64==0, K%32==0.
#define WMMA_BF16(ACC, AV, BV) \
  ACC = __builtin_amdgcn_wmma_f32_16x16x32_bf16(false, AV, false, BV, \
                                                (short)0, ACC, false, false)

template <int K>
__global__ void wmma_gemm_bf16_t(const unsigned short* __restrict__ A, int lda,
                                 const int* __restrict__ rowidx,
                                 const unsigned short* __restrict__ B,
                                 const float* __restrict__ bias,
                                 float* __restrict__ C, int ldc,
                                 int M, int N)
{
  const int lane = threadIdx.x & 31;
  const int wave = threadIdx.x >> 5;
  const int n4 = N >> 6;
  const int m2 = M >> 5;                       // 32-row groups
  const int gwave = blockIdx.x * (blockDim.x >> 5) + wave;
  if (gwave >= m2 * n4) return;
  const int mt = gwave / n4;
  const int ng = gwave - mt * n4;

  const int l15 = lane & 15;
  const int kg8 = (lane >> 4) << 3;            // 0 or 8: per-lane K sub-offset

  const int mrow0 = (mt << 5) + l15;
  const long rid0 = rowidx ? (long)rowidx[mrow0] : (long)mrow0;
  const long rid1 = rowidx ? (long)rowidx[mrow0 + 16] : (long)(mrow0 + 16);

  // Three live pointers, advanced 64B/iter; B rows 1..3 reached via
  // compile-time immediate offsets (16*K elements apart).
  const unsigned short* ap0 = A + rid0 * (long)lda + kg8;
  const unsigned short* ap1 = A + rid1 * (long)lda + kg8;
  const unsigned short* bp  = B + (long)((ng << 6) + l15) * K + kg8;

  v8f c00 = {}, c01 = {}, c02 = {}, c03 = {};
  v8f c10 = {}, c11 = {}, c12 = {}, c13 = {};

  for (int kk = 0; kk < K; kk += 32) {
    FragBF a0, a1, b0, b1, b2, b3;
    a0.q[0] = *(const uint4*)(ap0);
    a0.q[1] = *(const uint4*)(ap0 + 16);
    a1.q[0] = *(const uint4*)(ap1);
    a1.q[1] = *(const uint4*)(ap1 + 16);
    b0.q[0] = *(const uint4*)(bp);
    b0.q[1] = *(const uint4*)(bp + 16);
    b1.q[0] = *(const uint4*)(bp + 16 * K);
    b1.q[1] = *(const uint4*)(bp + 16 * K + 16);
    b2.q[0] = *(const uint4*)(bp + 32 * K);
    b2.q[1] = *(const uint4*)(bp + 32 * K + 16);
    b3.q[0] = *(const uint4*)(bp + 48 * K);
    b3.q[1] = *(const uint4*)(bp + 48 * K + 16);
    // prefetch next k-step of B: immediate offsets on the same base VGPRs
    __builtin_prefetch(bp + 32, 0, 1);
    __builtin_prefetch(bp + 32 * K + 32, 0, 1);
    WMMA_BF16(c00, a0.v, b0.v);
    WMMA_BF16(c10, a1.v, b0.v);
    WMMA_BF16(c01, a0.v, b1.v);
    WMMA_BF16(c11, a1.v, b1.v);
    WMMA_BF16(c02, a0.v, b2.v);
    WMMA_BF16(c12, a1.v, b2.v);
    WMMA_BF16(c03, a0.v, b3.v);
    WMMA_BF16(c13, a1.v, b3.v);
    ap0 += 32; ap1 += 32; bp += 32;
  }

  const int col0   = (ng << 6) + l15;
  const int rbase0 = (mt << 5) + ((lane >> 4) << 3);
  const int rbase1 = rbase0 + 16;
  const float bs0 = bias ? bias[col0]      : 0.0f;
  const float bs1 = bias ? bias[col0 + 16] : 0.0f;
  const float bs2 = bias ? bias[col0 + 32] : 0.0f;
  const float bs3 = bias ? bias[col0 + 48] : 0.0f;
#pragma unroll
  for (int r = 0; r < 8; ++r) {
    long row0 = rbase0 + r;
    long row1 = rbase1 + r;
    C[row0 * (long)ldc + col0]      = c00[r] + bs0;
    C[row0 * (long)ldc + col0 + 16] = c01[r] + bs1;
    C[row0 * (long)ldc + col0 + 32] = c02[r] + bs2;
    C[row0 * (long)ldc + col0 + 48] = c03[r] + bs3;
    C[row1 * (long)ldc + col0]      = c10[r] + bs0;
    C[row1 * (long)ldc + col0 + 16] = c11[r] + bs1;
    C[row1 * (long)ldc + col0 + 32] = c12[r] + bs2;
    C[row1 * (long)ldc + col0 + 48] = c13[r] + bs3;
  }
}

// ---------------------------------------------------------------------------
// GRU gate fusion.  Gi = x@Wih^T+bih, Gh = h@Whh^T+bhh (both [128,1536]).
//   r = sig(Gi_r+Gh_r); z = sig(Gi_z+Gh_z); n = tanh(Gi_n + r*Gh_n)
//   h' = (1-z)*n + z*h.  Writes fp32 h (in place) + bf16 into xh staging.
__global__ void gru_gates_k(const float* __restrict__ Gi, const float* __restrict__ Gh,
                            float* __restrict__ h, unsigned short* __restrict__ hbf,
                            int hbf_stride, float* __restrict__ enc_out_s)
{
  int idx = blockIdx.x * blockDim.x + threadIdx.x; // 128*512
  int b = idx >> 9, j = idx & 511;
  long base = (long)b * 1536 + j;
  float r  = 1.0f / (1.0f + __expf(-(Gi[base]        + Gh[base])));
  float z  = 1.0f / (1.0f + __expf(-(Gi[base + 512]  + Gh[base + 512])));
  float nn = tanhf(Gi[base + 1024] + r * Gh[base + 1024]);
  float hold = h[(long)b * 512 + j];
  float hnew = (1.0f - z) * nn + z * hold;
  h[(long)b * 512 + j] = hnew;
  hbf[(long)b * hbf_stride + j] = f2bf(hnew);
  if (enc_out_s) enc_out_s[(long)b * 512 + j] = hnew;
}

// ---------------------------------------------------------------------------
// e[s,b] = dot(q[b,:], enc_out[s,b,:]) + bil_b.  One wave per (s,b) pair.
__global__ void attn_scores_k(const float* __restrict__ q,
                              const float* __restrict__ enc_out,
                              const float* __restrict__ bil_b,
                              float* __restrict__ e)
{
  int gw   = (blockIdx.x * blockDim.x + threadIdx.x) >> 5;
  int lane = threadIdx.x & 31;
  if (gw >= 400 * 128) return;
  int s = gw >> 7, b = gw & 127;
  const float* eo = enc_out + ((long)s * 128 + b) * 512;
  const float* qb = q + (long)b * 512;
  float sum = 0.0f;
#pragma unroll
  for (int k = lane * 4; k < 512; k += 128) {
    float4 a = *(const float4*)(eo + k);
    float4 c = *(const float4*)(qb + k);
    sum += a.x * c.x + a.y * c.y + a.z * c.z + a.w * c.w;
  }
  for (int off = 16; off > 0; off >>= 1) sum += __shfl_down(sum, off, 32);
  if (lane == 0) e[(long)s * 128 + b] = sum + bil_b[0];
}

// softmax over s (400) for one batch row per block; writes attn [S,B] scratch
// and attn.T [B,S] directly into d_out slice for this decode step.
__global__ void attn_softmax_k(const float* __restrict__ e,
                               float* __restrict__ attn,
                               float* __restrict__ attn_out)
{
  __shared__ float red[256];
  int b = blockIdx.x, tid = threadIdx.x;
  float m = -INFINITY;
  for (int s = tid; s < 400; s += 256) m = fmaxf(m, e[(long)s * 128 + b]);
  red[tid] = m; __syncthreads();
  for (int w = 128; w > 0; w >>= 1) {
    if (tid < w) red[tid] = fmaxf(red[tid], red[tid + w]);
    __syncthreads();
  }
  m = red[0]; __syncthreads();
  float sum = 0.0f;
  for (int s = tid; s < 400; s += 256) sum += __expf(e[(long)s * 128 + b] - m);
  red[tid] = sum; __syncthreads();
  for (int w = 128; w > 0; w >>= 1) {
    if (tid < w) red[tid] += red[tid + w];
    __syncthreads();
  }
  float inv = 1.0f / red[0];
  for (int s = tid; s < 400; s += 256) {
    float a = __expf(e[(long)s * 128 + b] - m) * inv;
    attn[(long)s * 128 + b] = a;
    attn_out[(long)b * 400 + s] = a;
  }
}

// ctx[b,j] = sum_s attn[s,b]*enc_out[s,b,j]; writes bf16 into xh[:,512:1024]
__global__ void attn_ctx_k(const float* __restrict__ attn,
                           const float* __restrict__ enc_out,
                           unsigned short* __restrict__ xh)
{
  int idx = blockIdx.x * blockDim.x + threadIdx.x; // 128*512
  int b = idx >> 9, j = idx & 511;
  float acc = 0.0f;
  for (int s = 0; s < 400; ++s)
    acc += attn[(long)s * 128 + b] * enc_out[((long)s * 128 + b) * 512 + j];
  xh[(long)b * 1024 + 512 + j] = f2bf(acc);
}

// greedy argmax over 32000 logits per batch row (== argmax of log_softmax)
__global__ void row_argmax_k(const float* __restrict__ logits,
                             float* __restrict__ tok_out_f,
                             int* __restrict__ tok_buf)
{
  __shared__ float bv[256];
  __shared__ int   bi[256];
  int b = blockIdx.x, tid = threadIdx.x;
  float best = -INFINITY; int bidx = 0;
  for (int v = tid; v < 32000; v += 256) {
    float x = logits[(long)b * 32000 + v];
    if (x > best) { best = x; bidx = v; }
  }
  bv[tid] = best; bi[tid] = bidx; __syncthreads();
  for (int w = 128; w > 0; w >>= 1) {
    if (tid < w) {
      if (bv[tid + w] > bv[tid] ||
          (bv[tid + w] == bv[tid] && bi[tid + w] < bi[tid])) {
        bv[tid] = bv[tid + w]; bi[tid] = bi[tid + w];
      }
    }
    __syncthreads();
  }
  if (tid == 0) { tok_out_f[b] = (float)bi[0]; tok_buf[b] = bi[0]; }
}

// ---------------------------------------------------------------------------
extern "C" void kernel_launch(void* const* d_in, const int* in_sizes, int n_in,
                              void* d_out, int out_size, void* d_ws, size_t ws_size,
                              hipStream_t stream)
{
  const int   S = 400, B = 128, H = 512, V = 32000, T = 48, G3 = 1536;

  const int*   tok_in    = (const int*)  d_in[0];
  const float* embedding = (const float*)d_in[1];
  const float* enc_Wih   = (const float*)d_in[2];
  const float* enc_Whh   = (const float*)d_in[3];
  const float* enc_bih   = (const float*)d_in[4];
  const float* enc_bhh   = (const float*)d_in[5];
  const float* dec_Wih   = (const float*)d_in[6];
  const float* dec_Whh   = (const float*)d_in[7];
  const float* dec_bih   = (const float*)d_in[8];
  const float* dec_bhh   = (const float*)d_in[9];
  const float* bil_W     = (const float*)d_in[10];
  const float* bil_b     = (const float*)d_in[11];
  const float* out_W     = (const float*)d_in[12];
  const float* out_b     = (const float*)d_in[13];
  const int*   sos       = (const int*)  d_in[14];

  // ---- workspace carve-up (256B aligned slabs) ----
  char* wsp = (char*)d_ws;
  size_t off = 0;
  auto alloc = [&](size_t bytes) -> void* {
    void* p = wsp + off;
    off += (bytes + 255) & ~(size_t)255;
    return p;
  };
  unsigned short* emb_bf   = (unsigned short*)alloc((size_t)V * H * 2);
  unsigned short* encWih_b = (unsigned short*)alloc((size_t)G3 * H * 2);
  unsigned short* encWhh_b = (unsigned short*)alloc((size_t)G3 * H * 2);
  unsigned short* decWih_b = (unsigned short*)alloc((size_t)G3 * H * 2);
  unsigned short* decWhh_b = (unsigned short*)alloc((size_t)G3 * H * 2);
  unsigned short* bilWT_b  = (unsigned short*)alloc((size_t)H * H * 2);
  unsigned short* outW_b   = (unsigned short*)alloc((size_t)V * 2 * H * 2);
  float*          enc_out  = (float*)alloc((size_t)S * B * H * 4);
  float*          h_f32    = (float*)alloc((size_t)B * H * 4);
  unsigned short* xh_bf    = (unsigned short*)alloc((size_t)B * 2 * H * 2);
  float*          Gi       = (float*)alloc((size_t)B * G3 * 4);
  float*          Gh       = (float*)alloc((size_t)B * G3 * 4);
  float*          qbuf     = (float*)alloc((size_t)B * H * 4);
  float*          ebuf     = (float*)alloc((size_t)S * B * 4);
  float*          attn     = (float*)alloc((size_t)S * B * 4);
  float*          logits   = (float*)alloc((size_t)B * V * 4);
  int*            tok_buf  = (int*)alloc((size_t)B * 4);
  (void)ws_size; (void)in_sizes; (void)n_in; (void)out_size;

  // ---- one-time fp32 -> bf16 weight conversion ----
  auto cvt = [&](const float* src, unsigned short* dst, long n) {
    f32_to_bf16_k<<<(unsigned)((n + 255) / 256), 256, 0, stream>>>(src, dst, n);
  };
  cvt(embedding, emb_bf,   (long)V * H);
  cvt(enc_Wih,   encWih_b, (long)G3 * H);
  cvt(enc_Whh,   encWhh_b, (long)G3 * H);
  cvt(dec_Wih,   decWih_b, (long)G3 * H);
  cvt(dec_Whh,   decWhh_b, (long)G3 * H);
  cvt(out_W,     outW_b,   (long)V * 2 * H);
  transpose_bf16_k<<<(H * H) / 256, 256, 0, stream>>>(bil_W, bilWT_b);
  init_state_k<<<(B * 2 * H) / 256, 256, 0, stream>>>(h_f32, xh_bf, sos, tok_buf);

  // GEMM launch geometries: waves = (M/32)*(N/64), 8 waves / block
  const int blkG3 = (B / 32) * (G3 / 64) / 8;               // 12  (N=1536)
  const int blkH  = ((B / 32) * (H / 64) + 7) / 8;          // 4   (N=512)
  const int blkV  = (B / 32) * (V / 64) / 8;                // 250 (N=32000)

  // ---- encoder: 400 recurrent steps ----
  for (int s = 0; s < S; ++s) {
    // Gi = embedding[tok_s] @ enc_Wih^T + bih   (embedding gather fused as A)
    wmma_gemm_bf16_t<512><<<blkG3, 256, 0, stream>>>(
        emb_bf, H, tok_in + (long)s * B, encWih_b, enc_bih, Gi, G3, B, G3);
    // Gh = h @ enc_Whh^T + bhh   (A = xh staging rows, lda=1024)
    wmma_gemm_bf16_t<512><<<blkG3, 256, 0, stream>>>(
        xh_bf, 2 * H, nullptr, encWhh_b, enc_bhh, Gh, G3, B, G3);
    gru_gates_k<<<(B * H) / 256, 256, 0, stream>>>(Gi, Gh, h_f32, xh_bf, 2 * H,
                                                   enc_out + (long)s * B * H);
  }

  // ---- decoder: 48 greedy steps ----
  float* tok_out  = (float*)d_out;              // [T,B] token ids (as float)
  float* attn_out = (float*)d_out + (long)T * B;// [T,B,S]
  for (int t = 0; t < T; ++t) {
    wmma_gemm_bf16_t<512><<<blkG3, 256, 0, stream>>>(
        emb_bf, H, tok_buf, decWih_b, dec_bih, Gi, G3, B, G3);
    wmma_gemm_bf16_t<512><<<blkG3, 256, 0, stream>>>(
        xh_bf, 2 * H, nullptr, decWhh_b, dec_bhh, Gh, G3, B, G3);
    gru_gates_k<<<(B * H) / 256, 256, 0, stream>>>(Gi, Gh, h_f32, xh_bf, 2 * H,
                                                   nullptr);
    // q = h @ bil_W  (bilWT_b is W^T in [N,K] form)
    wmma_gemm_bf16_t<512><<<blkH, 256, 0, stream>>>(
        xh_bf, 2 * H, nullptr, bilWT_b, nullptr, qbuf, H, B, H);
    attn_scores_k<<<(S * B * 32) / 256, 256, 0, stream>>>(qbuf, enc_out, bil_b, ebuf);
    attn_softmax_k<<<B, 256, 0, stream>>>(ebuf, attn,
                                          attn_out + (long)t * B * S);
    attn_ctx_k<<<(B * H) / 256, 256, 0, stream>>>(attn, enc_out, xh_bf);
    // logits = [h,ctx] @ out_W^T + out_b  (the big, L2-resident GEMM)
    wmma_gemm_bf16_t<1024><<<blkV, 256, 0, stream>>>(
        xh_bf, 2 * H, nullptr, outW_b, out_b, logits, V, B, V);
    row_argmax_k<<<B, 256, 0, stream>>>(logits, tok_out + (long)t * B, tok_buf);
  }
}
